// ExpertTransformerCommon_82695300317490
// MI455X (gfx1250) — compile-verified, matrix operand order checked by reading, not run on hardware
//
#include <hip/hip_runtime.h>
#include <hip/hip_bf16.h>
#include <math.h>

#define BB   8
#define SS   1024
#define DD   768
#define HH   12
#define DHH  64
#define FFD  3072
#define EE   8
#define TT   (BB*SS)          // 8192 tokens
#define CAPN (2*TT/EE)        // 2048 capacity per expert

typedef __attribute__((ext_vector_type(16))) __bf16 bf16x16;
typedef __attribute__((ext_vector_type(8)))  __bf16 bf16x8;
typedef __attribute__((ext_vector_type(8)))  float  f32x8;

// ---------- scalar helpers ----------
__device__ __forceinline__ __bf16 f2bf(float f) { return (__bf16)f; }

__device__ __forceinline__ float geluf(float x) {
    return 0.5f * x * (1.0f + erff(x * 0.70710678118654752f));
}

// ---------- WMMA wrapper ----------
__device__ __forceinline__ f32x8 wmma_bf16(bf16x16 a, bf16x16 b, f32x8 c) {
    return __builtin_amdgcn_wmma_f32_16x16x32_bf16(false, a, false, b,
                                                   (short)0, c, false, false);
}

// Load a 16x32 bf16 fragment (A-layout / transposed-B layout) from an LDS row.
// Row chunk layout: elements [hi*8 .. hi*8+7] then [16+hi*8 .. 16+hi*8+7].
__device__ __forceinline__ bf16x16 load_frag(const __bf16* rp, int hi) {
    bf16x8 lo = *(const bf16x8*)(rp + hi * 8);
    bf16x8 hv = *(const bf16x8*)(rp + 16 + hi * 8);
    bf16x16 r;
#pragma unroll
    for (int i = 0; i < 8; i++) { r[i] = lo[i]; r[8 + i] = hv[i]; }
    return r;
}

// ---------- Tensor Data Mover (TDM) 2D tile load: global bf16 -> LDS ----------
#if __has_builtin(__builtin_amdgcn_tensor_load_to_lds) && __has_builtin(__builtin_amdgcn_s_wait_tensorcnt)
#define HAVE_TDM 1
#else
#define HAVE_TDM 0
#endif

#define TDM_STR2(x) #x
#define TDM_STR(x) TDM_STR2(x)
#if HAVE_TDM
#pragma message("CDNA5 probe: TDM tensor_load_to_lds path ENABLED (clang " TDM_STR(__clang_major__) ")")
#else
#pragma message("CDNA5 probe: TDM builtins NOT available (clang " TDM_STR(__clang_major__) "); manual LDS staging")
#endif

#if HAVE_TDM
typedef __attribute__((ext_vector_type(4))) unsigned int u32x4;
typedef __attribute__((ext_vector_type(4))) int i32x4;
typedef __attribute__((ext_vector_type(8))) int i32x8;

// D# per CDNA5 ISA: group0 = {count/lds_addr/global_addr/type},
// group1 = {data_size, tensor_dim0/1, tile_dim0/1, tensor_dim0_stride}. 2-byte elems.
__device__ __forceinline__ void tdm_load_tile_2d_bf16(
        unsigned lds_off, const void* gptr,
        unsigned tensor_w, unsigned tensor_h,
        unsigned tile_w, unsigned tile_h,
        unsigned long long row_stride_elems)
{
    unsigned long long ga = (unsigned long long)gptr;
    u32x4 g0;
    g0[0] = 1u;                                       // count=1, user descriptor
    g0[1] = lds_off;                                  // lds_addr (bytes)
    g0[2] = (unsigned)(ga & 0xffffffffu);             // global_addr[31:0]
    g0[3] = (unsigned)((ga >> 32) & 0x1ffffffu)       // global_addr[56:32]
          | 0x80000000u;                              // type=2 ("image") bits 127:126
    i32x8 g1;
    g1[0] = (int)(1u << 16);                          // workgroup_mask=0, data_size=1 (2B)
    g1[1] = (int)((tensor_w & 0xffffu) << 16);        // tensor_dim0[15:0]
    g1[2] = (int)((tensor_w >> 16) | ((tensor_h & 0xffffu) << 16));
    g1[3] = (int)((tensor_h >> 16) | (tile_w << 16)); // tile_dim0
    g1[4] = (int)tile_h;                              // tile_dim1 (tile_dim2=0)
    g1[5] = (int)(unsigned)(row_stride_elems & 0xffffffffu);
    g1[6] = (int)((unsigned)(row_stride_elems >> 32) & 0xffffu);
    g1[7] = 0;
    i32x4 z = {0, 0, 0, 0};
#if __clang_major__ >= 23
    i32x8 z8 = {0, 0, 0, 0, 0, 0, 0, 0};
    __builtin_amdgcn_tensor_load_to_lds(g0, g1, z, z, z8, 0);
#else
    __builtin_amdgcn_tensor_load_to_lds(g0, g1, z, z, 0);
#endif
}
#endif

// =====================================================================
// Generic tiled WMMA GEMM: C[M,N] = A[M,K] * W[K,N] (+bias, +gelu)
// A: fp32 or bf16; W/bias fp32; C: fp32 or bf16. Batched via blockIdx.z.
// Block = 256 threads (8 waves); block tile 64x128; wave tile 16x64.
// Double-buffered LDS ping-pong: tile i+1 staged (TDM for bf16 A) while
// WMMAs consume tile i; one barrier per K step.
// =====================================================================
template<int A_BF16, int OUT_BF16, int EPI>
__global__ __launch_bounds__(256)
void gemm_wmma(const void* __restrict__ Av, const float* __restrict__ W,
               const float* __restrict__ bias, void* __restrict__ Cv,
               int M, int N, int Kd,
               long aBatch, long wBatch, long biasBatch, long cBatch)
{
    __shared__ __bf16 ldsA[2][64][32];
    __shared__ __bf16 ldsB[2][128][32];   // stored [n][k]

    const int batch = blockIdx.z;
    const float* Wb = W + (long)batch * wBatch;
    const float* Bb = bias + (long)batch * biasBatch;

    const int tid  = threadIdx.x;
    const int wave = tid >> 5, lane = tid & 31;
    const int hi = lane >> 4, ln = lane & 15;
    const int mt = wave & 3;      // 0..3 -> m sub-tile (16 rows)
    const int nh = wave >> 2;     // 0..1 -> n half (64 wide)
    const int mBase = blockIdx.y * 64, nBase = blockIdx.x * 128;

    const int arow = tid >> 2, acol = (tid & 3) * 8;   // A tile 64x32, 8 elems/thr
    const int bkr  = tid >> 3, bnc  = (tid & 7) * 16;  // W tile 32x128, 16 elems/thr

    // stage one K-step tile (A + W) into ping-pong buffer pb
    auto stage = [&](int kk, int pb) {
        if (A_BF16) {
            const __bf16* A = ((const __bf16*)Av) + (long)batch * aBatch;
#if HAVE_TDM
            if (wave == 0) {
                tdm_load_tile_2d_bf16((unsigned)(unsigned long long)&ldsA[pb][0][0],
                                      A + (long)mBase * Kd + kk,
                                      (unsigned)Kd, (unsigned)M, 32u, 64u,
                                      (unsigned long long)Kd);
            }
#else
            const __bf16* asrc = A + (long)(mBase + arow) * Kd + kk + acol;
#pragma unroll
            for (int i = 0; i < 8; i++) ldsA[pb][arow][acol + i] = asrc[i];
#endif
        } else {
            const float* A = ((const float*)Av) + (long)batch * aBatch;
            const float* asrc = A + (long)(mBase + arow) * Kd + kk + acol;
#pragma unroll
            for (int i = 0; i < 8; i++) ldsA[pb][arow][acol + i] = f2bf(asrc[i]);
        }
        const float* wsrc = Wb + (long)(kk + bkr) * N + nBase + bnc;
#pragma unroll
        for (int i = 0; i < 16; i++) ldsB[pb][bnc + i][bkr] = f2bf(wsrc[i]);
    };

    f32x8 acc[4] = {};

    stage(0, 0);
#if HAVE_TDM
    if (A_BF16 && wave == 0) __builtin_amdgcn_s_wait_tensorcnt(0);
#endif
    __syncthreads();

    int pb = 0;
    for (int kk = 0; kk < Kd; kk += 32) {
        // issue staging of next tile into the other buffer (runs under compute)
        if (kk + 32 < Kd) {
            stage(kk + 32, pb ^ 1);
            // prefetch tile i+2 of W (global_prefetch_b8 on gfx1250)
            if (kk + 64 < Kd)
                __builtin_prefetch(Wb + (long)(kk + 64 + bkr) * N + nBase + bnc, 0, 1);
        }

        const bf16x16 af = load_frag(&ldsA[pb][mt * 16 + ln][0], hi);
#pragma unroll
        for (int s2 = 0; s2 < 4; s2++) {
            bf16x16 bfr = load_frag(&ldsB[pb][nh * 64 + s2 * 16 + ln][0], hi);
            acc[s2] = wmma_bf16(af, bfr, acc[s2]);
        }

#if HAVE_TDM
        if (A_BF16 && wave == 0) __builtin_amdgcn_s_wait_tensorcnt(0);
#endif
        __syncthreads();
        pb ^= 1;
    }

    // ---- epilogue: bias / gelu / store ----
#pragma unroll
    for (int s2 = 0; s2 < 4; s2++) {
#pragma unroll
        for (int r = 0; r < 8; r++) {
            const int m = mBase + mt * 16 + hi * 8 + r;
            const int n = nBase + nh * 64 + s2 * 16 + ln;
            float v = acc[s2][r];
            if (EPI >= 1) v += Bb[n];
            if (EPI == 2) v = geluf(v);
            if (OUT_BF16) {
                __bf16* C = ((__bf16*)Cv) + (long)batch * cBatch;
                C[(long)m * N + n] = f2bf(v);
            } else {
                float* C = ((float*)Cv) + (long)batch * cBatch;
                C[(long)m * N + n] = v;
            }
        }
    }
}

// =====================================================================
// Flash attention: 128 threads (4 waves); block = (b, h, 64 query rows).
// Each wave owns 16 query rows; online softmax; QK^T and PV via WMMA.
// =====================================================================
__global__ __launch_bounds__(128)
void attn_kernel(const __bf16* __restrict__ Q, const __bf16* __restrict__ K,
                 const __bf16* __restrict__ V, const float* __restrict__ mask,
                 __bf16* __restrict__ ctx)
{
    __shared__ __bf16 lQ[64][64];        // query rows x 64 feats
    __shared__ __bf16 lK[32][64];        // key tile x feats (B-layout rows = keys)
    __shared__ __bf16 lV[64][32];        // feat x key (transposed V tile)
    __shared__ __bf16 lP[4][16][32];     // per-wave probs relayout
    __shared__ float  lM[32];

    const int blk = blockIdx.x;
    const int qb = blk % (SS / 64);
    const int h  = (blk / (SS / 64)) % HH;
    const int b  = blk / ((SS / 64) * HH);

    const int tid = threadIdx.x;
    const int w = tid >> 5, lane = tid & 31;
    const int hi = lane >> 4, ln = lane & 15;

    { // stage Q block
        const int row = tid >> 1, f0 = (tid & 1) * 32;
        const __bf16* src = Q + ((long)(b * SS + qb * 64 + row)) * DD + h * DHH + f0;
#pragma unroll
        for (int i = 0; i < 32; i++) lQ[row][f0 + i] = src[i];
    }
    __syncthreads();

    f32x8 o0 = {}, o1 = {}, o2 = {}, o3 = {};
    float mi[8], li[8];
#pragma unroll
    for (int r = 0; r < 8; r++) { mi[r] = -1e30f; li[r] = 0.f; }

    const bf16x16 q0 = load_frag(&lQ[w * 16 + ln][0], hi);
    const bf16x16 q1 = load_frag(&lQ[w * 16 + ln][32], hi);

    for (int kt = 0; kt < SS / 32; kt++) {
        // ---- stage K tile 32x64 (rows = keys): TDM strided 2D tile ----
#if HAVE_TDM
        if (w == 0) {
            tdm_load_tile_2d_bf16((unsigned)(unsigned long long)&lK[0][0],
                                  K + ((long)(b * SS + kt * 32)) * DD + h * DHH,
                                  (unsigned)DD, (unsigned)TT, 64u, 32u,
                                  (unsigned long long)DD);
            __builtin_amdgcn_s_wait_tensorcnt(0);
        }
#else
        {
            const int row = tid >> 2, f0 = (tid & 3) * 16;
            const __bf16* src = K + ((long)(b * SS + kt * 32 + row)) * DD + h * DHH + f0;
#pragma unroll
            for (int i = 0; i < 16; i++) lK[row][f0 + i] = src[i];
        }
#endif
        { // stage V tile transposed -> lV[feat][key]
            const int kloc = tid >> 2, f0 = (tid & 3) * 16;
            const __bf16* src = V + ((long)(b * SS + kt * 32 + kloc)) * DD + h * DHH + f0;
#pragma unroll
            for (int i = 0; i < 16; i++) lV[f0 + i][kloc] = src[i];
        }
        if (tid < 32) lM[tid] = mask[b * SS + kt * 32 + tid];
        __syncthreads();

        // ---- scores: two 16x16 tiles (keys ln and 16+ln) ----
        f32x8 s0 = {}, s1 = {};
        {
            bf16x16 k0a = load_frag(&lK[ln][0], hi);
            bf16x16 k0b = load_frag(&lK[ln][32], hi);
            bf16x16 k1a = load_frag(&lK[16 + ln][0], hi);
            bf16x16 k1b = load_frag(&lK[16 + ln][32], hi);
            s0 = wmma_bf16(q0, k0a, s0); s0 = wmma_bf16(q1, k0b, s0);
            s1 = wmma_bf16(q0, k1a, s1); s1 = wmma_bf16(q1, k1b, s1);
        }

        // ---- online softmax (rows hi*8+r live in this lane's 16-group) ----
        float p0[8], p1[8];
#pragma unroll
        for (int r = 0; r < 8; r++) {
            float a   = s0[r] * 0.125f + lM[ln];
            float bb2 = s1[r] * 0.125f + lM[16 + ln];
            float mx = fmaxf(a, bb2);
#pragma unroll
            for (int d = 1; d < 16; d <<= 1) mx = fmaxf(mx, __shfl_xor(mx, d, 32));
            float nm = fmaxf(mi[r], mx);
            float sc = __expf(mi[r] - nm);
            float e0 = __expf(a - nm), e1 = __expf(bb2 - nm);
            float rs = e0 + e1;
#pragma unroll
            for (int d = 1; d < 16; d <<= 1) rs += __shfl_xor(rs, d, 32);
            li[r] = li[r] * sc + rs;
            mi[r] = nm;
            o0[r] *= sc; o1[r] *= sc; o2[r] *= sc; o3[r] *= sc;
            p0[r] = e0;  p1[r] = e1;
        }

        // relayout P (C-layout -> A-layout) through LDS
#pragma unroll
        for (int r = 0; r < 8; r++) {
            lP[w][hi * 8 + r][ln]      = f2bf(p0[r]);
            lP[w][hi * 8 + r][16 + ln] = f2bf(p1[r]);
        }
        __syncthreads();

        // ---- PV: P(16x32) x V(32x64) ----
        {
            bf16x16 ap  = load_frag(&lP[w][ln][0], hi);
            bf16x16 v0f = load_frag(&lV[ln][0], hi);
            bf16x16 v1f = load_frag(&lV[16 + ln][0], hi);
            bf16x16 v2f = load_frag(&lV[32 + ln][0], hi);
            bf16x16 v3f = load_frag(&lV[48 + ln][0], hi);
            o0 = wmma_bf16(ap, v0f, o0);
            o1 = wmma_bf16(ap, v1f, o1);
            o2 = wmma_bf16(ap, v2f, o2);
            o3 = wmma_bf16(ap, v3f, o3);
        }
        __syncthreads();
    }

#pragma unroll
    for (int r = 0; r < 8; r++) {
        const int q = qb * 64 + w * 16 + hi * 8 + r;
        const float inv = 1.0f / li[r];
        const long base = ((long)(b * SS + q)) * DD + h * DHH;
        ctx[base + ln]      = f2bf(o0[r] * inv);
        ctx[base + 16 + ln] = f2bf(o1[r] * inv);
        ctx[base + 32 + ln] = f2bf(o2[r] * inv);
        ctx[base + 48 + ln] = f2bf(o3[r] * inv);
    }
}

// =====================================================================
// out = LayerNorm(x + y) * g + b   (one block per 768-wide row)
// =====================================================================
__global__ __launch_bounds__(256)
void add_ln_kernel(const float* __restrict__ x, const float* __restrict__ y,
                   const float* __restrict__ g, const float* __restrict__ bb,
                   float* __restrict__ out)
{
    __shared__ float red[256];
    const long row = blockIdx.x;
    const int tid = threadIdx.x;
    const float* xr = x + row * DD;
    const float* yr = y + row * DD;
    float v[3]; float s = 0.f;
#pragma unroll
    for (int i = 0; i < 3; i++) { v[i] = xr[tid + i * 256] + yr[tid + i * 256]; s += v[i]; }
    red[tid] = s; __syncthreads();
    for (int st = 128; st > 0; st >>= 1) { if (tid < st) red[tid] += red[tid + st]; __syncthreads(); }
    const float mu = red[0] / DD;
    __syncthreads();
    float s2 = 0.f;
#pragma unroll
    for (int i = 0; i < 3; i++) { float d = v[i] - mu; s2 += d * d; }
    red[tid] = s2; __syncthreads();
    for (int st = 128; st > 0; st >>= 1) { if (tid < st) red[tid] += red[tid + st]; __syncthreads(); }
    const float inv = rsqrtf(red[0] / DD + 1e-12f);
#pragma unroll
    for (int i = 0; i < 3; i++) {
        const int c = tid + i * 256;
        out[row * DD + c] = (v[i] - mu) * inv * g[c] + bb[c];
    }
}

// =====================================================================
// Router: logits = xf @ Wr + br; softmax; top-1 gate + expert index.
// =====================================================================
__global__ __launch_bounds__(256)
void router_kernel(const float* __restrict__ xf, const float* __restrict__ Wr,
                   const float* __restrict__ br, int* __restrict__ eidx,
                   float* __restrict__ gate)
{
    const int t = blockIdx.x * blockDim.x + threadIdx.x;
    if (t >= TT) return;
    float acc[EE];
#pragma unroll
    for (int e = 0; e < EE; e++) acc[e] = br[e];
    const float* xr = xf + (long)t * DD;
    for (int d = 0; d < DD; d++) {
        const float x = xr[d];
#pragma unroll
        for (int e = 0; e < EE; e++) acc[e] += x * Wr[d * EE + e];
    }
    float mx = acc[0]; int ai = 0;
#pragma unroll
    for (int e = 1; e < EE; e++) if (acc[e] > mx) { mx = acc[e]; ai = e; }
    float sum = 0.f;
#pragma unroll
    for (int e = 0; e < EE; e++) sum += __expf(acc[e] - mx);
    gate[t] = 1.0f / sum;   // softmax prob of argmax
    eidx[t] = ai;
}

// Deterministic capacity scan: pos within expert, keep mask. Single block.
__global__ __launch_bounds__(256)
void scan_kernel(const int* __restrict__ eidx, int* __restrict__ pos,
                 int* __restrict__ keep)
{
    __shared__ int cnt[256][EE];
    const int tid = threadIdx.x;
    const int chunk = TT / 256;
    const int t0 = tid * chunk;
    int c[EE];
#pragma unroll
    for (int e = 0; e < EE; e++) c[e] = 0;
    for (int i = 0; i < chunk; i++) c[eidx[t0 + i]]++;
#pragma unroll
    for (int e = 0; e < EE; e++) cnt[tid][e] = c[e];
    __syncthreads();
    if (tid < EE) {
        int run = 0;
        for (int j = 0; j < 256; j++) { int tmp = cnt[j][tid]; cnt[j][tid] = run; run += tmp; }
    }
    __syncthreads();
#pragma unroll
    for (int e = 0; e < EE; e++) c[e] = cnt[tid][e];
    for (int i = 0; i < chunk; i++) {
        const int e = eidx[t0 + i];
        const int p = c[e]++;
        keep[t0 + i] = (p < CAPN) ? 1 : 0;
        pos[t0 + i]  = (p < CAPN) ? p : (CAPN - 1);
    }
}

__global__ void zero_bf16(__bf16* p, long n) {
    long i = (long)blockIdx.x * blockDim.x + threadIdx.x;
    const long stride = (long)gridDim.x * blockDim.x;
    const __bf16 z = (__bf16)0.f;
    for (; i < n; i += stride) p[i] = z;
}

__global__ __launch_bounds__(256)
void scatter_kernel(const float* __restrict__ xf, const int* __restrict__ eidx,
                    const int* __restrict__ pos, const int* __restrict__ keep,
                    __bf16* __restrict__ buf)
{
    const int t = blockIdx.x;
    if (!keep[t]) return;
    const long dst = ((long)eidx[t] * CAPN + pos[t]) * DD;
    const int tid = threadIdx.x;
#pragma unroll
    for (int i = 0; i < 3; i++)
        buf[dst + tid + i * 256] = f2bf(xf[(long)t * DD + tid + i * 256]);
}

__global__ __launch_bounds__(256)
void gather_kernel(const float* __restrict__ y, const int* __restrict__ eidx,
                   const int* __restrict__ pos, const int* __restrict__ keep,
                   const float* __restrict__ gate, float* __restrict__ ffn)
{
    const int t = blockIdx.x;
    const float gsc = keep[t] ? gate[t] : 0.f;
    const long src = ((long)eidx[t] * CAPN + pos[t]) * DD;
    const int tid = threadIdx.x;
#pragma unroll
    for (int i = 0; i < 3; i++)
        ffn[(long)t * DD + tid + i * 256] = y[src + tid + i * 256] * gsc;
}

// =====================================================================
extern "C" void kernel_launch(void* const* d_in, const int* in_sizes, int n_in,
                              void* d_out, int out_size, void* d_ws, size_t ws_size,
                              hipStream_t stream)
{
    (void)in_sizes; (void)n_in; (void)out_size; (void)ws_size;
    const float* hidden = (const float*)d_in[0];
    const float* amask  = (const float*)d_in[1];
    const float* Wq = (const float*)d_in[2];  const float* bq = (const float*)d_in[3];
    const float* Wk = (const float*)d_in[4];  const float* bk = (const float*)d_in[5];
    const float* Wv = (const float*)d_in[6];  const float* bv = (const float*)d_in[7];
    const float* Wo = (const float*)d_in[8];  const float* bo = (const float*)d_in[9];
    const float* ln1g = (const float*)d_in[10]; const float* ln1b = (const float*)d_in[11];
    const float* Wr = (const float*)d_in[12]; const float* br = (const float*)d_in[13];
    const float* W1 = (const float*)d_in[14]; const float* b1 = (const float*)d_in[15];
    const float* W2 = (const float*)d_in[16]; const float* b2 = (const float*)d_in[17];
    const float* ln2g = (const float*)d_in[18]; const float* ln2b = (const float*)d_in[19];
    float* out = (float*)d_out;

    char* ws = (char*)d_ws;
    auto alloc = [&](size_t bytes) -> char* {
        char* p = ws; ws += (bytes + 255) & ~(size_t)255; return p;
    };
    __bf16* Qb   = (__bf16*)alloc((size_t)TT * DD * 2);
    __bf16* Kb   = (__bf16*)alloc((size_t)TT * DD * 2);
    __bf16* Vb   = (__bf16*)alloc((size_t)TT * DD * 2);
    __bf16* ctxb = (__bf16*)alloc((size_t)TT * DD * 2);
    float*  proj = (float*)alloc((size_t)TT * DD * 4);    // reused as ffn output
    float*  atto = (float*)alloc((size_t)TT * DD * 4);
    __bf16* ebuf = (__bf16*)alloc((size_t)EE * CAPN * DD * 2);
    __bf16* hbuf = (__bf16*)alloc((size_t)EE * CAPN * FFD * 2);
    float*  ybuf = (float*)alloc((size_t)EE * CAPN * DD * 4);
    int*    eidx = (int*)alloc((size_t)TT * 4);
    int*    pos  = (int*)alloc((size_t)TT * 4);
    int*    keep = (int*)alloc((size_t)TT * 4);
    float*  gate = (float*)alloc((size_t)TT * 4);
    float*  ffn  = proj;

    const dim3 blk(256);

    // ---- QKV projections (fp32 in, bf16 out, +bias) ----
    gemm_wmma<0,1,1><<<dim3(DD/128, TT/64, 1), blk, 0, stream>>>(
        hidden, Wq, bq, Qb, TT, DD, DD, 0, 0, 0, 0);
    gemm_wmma<0,1,1><<<dim3(DD/128, TT/64, 1), blk, 0, stream>>>(
        hidden, Wk, bk, Kb, TT, DD, DD, 0, 0, 0, 0);
    gemm_wmma<0,1,1><<<dim3(DD/128, TT/64, 1), blk, 0, stream>>>(
        hidden, Wv, bv, Vb, TT, DD, DD, 0, 0, 0, 0);

    // ---- flash attention ----
    attn_kernel<<<dim3(BB * HH * (SS / 64)), dim3(128), 0, stream>>>(
        Qb, Kb, Vb, amask, ctxb);

    // ---- output projection (bf16 in via TDM, fp32 out, +bias) ----
    gemm_wmma<1,0,1><<<dim3(DD/128, TT/64, 1), blk, 0, stream>>>(
        ctxb, Wo, bo, proj, TT, DD, DD, 0, 0, 0, 0);

    // ---- att_output = LN(x + proj) ----
    add_ln_kernel<<<dim3(TT), blk, 0, stream>>>(hidden, proj, ln1g, ln1b, atto);

    // ---- router + deterministic capacity scan ----
    router_kernel<<<dim3((TT + 255) / 256), blk, 0, stream>>>(atto, Wr, br, eidx, gate);
    scan_kernel<<<dim3(1), blk, 0, stream>>>(eidx, pos, keep);

    // ---- scatter into per-expert buffers (bf16) ----
    zero_bf16<<<dim3(2048), blk, 0, stream>>>(ebuf, (long)EE * CAPN * DD);
    scatter_kernel<<<dim3(TT), blk, 0, stream>>>(atto, eidx, pos, keep, ebuf);

    // ---- expert FFN: h = gelu(buf @ W1 + b1); y = h @ W2 + b2 ----
    gemm_wmma<1,1,2><<<dim3(FFD/128, CAPN/64, EE), blk, 0, stream>>>(
        ebuf, W1, b1, hbuf, CAPN, FFD, DD,
        (long)CAPN * DD, (long)DD * FFD, (long)FFD, (long)CAPN * FFD);
    gemm_wmma<1,0,1><<<dim3(DD/128, CAPN/64, EE), blk, 0, stream>>>(
        hbuf, W2, b2, ybuf, CAPN, DD, FFD,
        (long)CAPN * FFD, (long)FFD * DD, (long)DD, (long)CAPN * DD);

    // ---- gather * gate; final LN into d_out ----
    gather_kernel<<<dim3(TT), blk, 0, stream>>>(ybuf, eidx, pos, keep, gate, ffn);
    add_ln_kernel<<<dim3(TT), blk, 0, stream>>>(atto, ffn, ln2g, ln2b, out);
}
